// RGCN_81784767250769
// MI455X (gfx1250) — matrix operand order repaired and checked by reading, not compiled, not run on hardware
//
#include <hip/hip_runtime.h>
#include <hip/hip_bf16.h>
#include <math.h>

// ---------------------------------------------------------------------------
// RGCN (2-layer, basis decomposition) for MI455X / gfx1250.
//
//   Layer: T[dst,b,:] += comp[rel,b]*norm*h[src,:]   (f32 atomics, scatter)
//          U  = bf16([T | h])          N x 2304
//          Wc = bf16([V_b ; loopw])    2304 x 256  (packed to WMMA frag order)
//          out = act(U @ Wc + bias)    via v_wmma_f32_16x16x32_bf16
//
// GEMM: block = 8 waves = 128 rows x 256 cols. B panel staged through LDS
// (double-buffered 2x16KB via global_load_async_to_lds_b128 / ASYNCcnt).
// B fragments pipelined 3-deep through 4 disjoint register groups (no copies,
// so regalloc cannot re-serialize); A fragments prefetched one k-step ahead
// in alternating registers.
// ---------------------------------------------------------------------------

#define HD   256
#define NB   8
#define KTOT (NB * HD + HD)   // 2304
#define KSTEPS (KTOT / 32)    // 72 (even)

typedef __attribute__((ext_vector_type(16))) __bf16        v16bf;
typedef __attribute__((ext_vector_type(8)))  float         v8f;
typedef __attribute__((ext_vector_type(4)))  unsigned int  v4u;
typedef int v4i_vs __attribute__((vector_size(16)));   // matches builtin "V4i"

union Frag32B { v16bf bf; v4u u[2]; };

#if defined(__has_builtin)
#if __has_builtin(__builtin_amdgcn_global_load_async_to_lds_b128)
#define HAVE_ASYNC_LDS 1
#endif
#endif

__device__ __forceinline__ void wait_asynccnt0() {
#if defined(__has_builtin) && __has_builtin(__builtin_amdgcn_s_wait_asynccnt)
    __builtin_amdgcn_s_wait_asynccnt(0);
#else
    asm volatile("s_wait_asynccnt 0x0" ::: "memory");
#endif
}

__device__ __forceinline__ unsigned short f32_to_bf16_rne(float f) {
    unsigned int u = __float_as_uint(f);
    u += 0x7FFFu + ((u >> 16) & 1u);          // round-to-nearest-even
    return (unsigned short)(u >> 16);
}

// ---------------------------------------------------------------- h0 gather
__global__ void k_gather(const int* __restrict__ node_id,
                         const float* __restrict__ emb,
                         float* __restrict__ h0, int n_nodes) {
    long idx = (long)blockIdx.x * blockDim.x + threadIdx.x;
    if (idx >= (long)n_nodes * HD) return;
    int n = (int)(idx >> 8);
    int i = (int)(idx & 255);
    h0[idx] = emb[(long)node_id[n] * HD + i];
}

// ---------------------------------------------------------------- zero ws
__global__ void k_zero(float4* __restrict__ p, long n4) {
    long i      = (long)blockIdx.x * blockDim.x + threadIdx.x;
    long stride = (long)gridDim.x * blockDim.x;
    float4 z = make_float4(0.f, 0.f, 0.f, 0.f);
    for (; i < n4; i += stride) p[i] = z;
}

// ----------------------------------------------------- per-edge scatter-add
// One wave per edge; lane handles 8 of 256 feature columns; 8 basis atomics.
__global__ void k_scatter(const int* __restrict__ rel,
                          const int* __restrict__ src,
                          const int* __restrict__ dst,
                          const float* __restrict__ norm,
                          const float* __restrict__ h,
                          const float* __restrict__ comp,
                          float* __restrict__ T, int n_edges) {
    int wid  = threadIdx.x >> 5;
    int lane = threadIdx.x & 31;
    int e = blockIdx.x * 8 + wid;
    if (e >= n_edges) return;
    int   r  = rel[e];
    int   s  = src[e];
    int   d  = dst[e];
    float nm = norm[e];
    float c[NB];
#pragma unroll
    for (int b = 0; b < NB; ++b) c[b] = comp[r * NB + b] * nm;
    const float* hs = h + (long)s * HD;
    float*       Td = T + (long)d * (NB * HD);
#pragma unroll
    for (int it = 0; it < HD / 32; ++it) {
        int   i = lane + 32 * it;
        float v = hs[i];
#pragma unroll
        for (int b = 0; b < NB; ++b)
            atomicAdd(&Td[b * HD + i], c[b] * v);   // global_atomic_add_f32
    }
}

// ----------------------------------------------- pack U = bf16([T | h])
__global__ void k_packU(const float* __restrict__ T,
                        const float* __restrict__ h,
                        unsigned short* __restrict__ U, int n_nodes) {
    long idx = (long)blockIdx.x * blockDim.x + threadIdx.x;
    if (idx >= (long)n_nodes * KTOT) return;
    int n = (int)(idx / KTOT);
    int k = (int)(idx % KTOT);
    float v = (k < NB * HD) ? T[(long)n * (NB * HD) + k]
                            : h[(long)n * HD + (k - NB * HD)];
    U[idx] = f32_to_bf16_rne(v);
}

// --------------------------------- pack Wc into WMMA B-fragment order (bf16)
// WF[((kt*16+nt)*32 + lane)*16 + j] = Wc[kt*32 + (lane>>4)*16 + j][nt*16 + (lane&15)]
// where Wc rows 0..2047 = bases (flat k*256+col), rows 2048..2303 = loopw.
__global__ void k_packW(const float* __restrict__ bases,
                        const float* __restrict__ loopw,
                        unsigned short* __restrict__ WF) {
    int idx = blockIdx.x * blockDim.x + threadIdx.x;   // < KTOT*HD = 589824
    int j    = idx & 15;
    int l    = (idx >> 4) & 31;
    int nt   = (idx >> 9) & 15;
    int kt   = idx >> 13;
    int half = l >> 4;
    int k    = kt * 32 + half * 16 + j;
    int col  = nt * 16 + (l & 15);
    float v = (k < NB * HD) ? bases[(long)k * HD + col]
                            : loopw[(long)(k - NB * HD) * HD + col];
    WF[idx] = f32_to_bf16_rne(v);
}

// ------------------- one k-step of WMMAs, B fragments pipelined 3-deep.
// 4 disjoint fragment buffers indexed by constants after full unroll: no
// register moves for regalloc to coalesce, so loads stay hoisted ~3 WMMAs
// ahead of their consumer (waits become s_wait_dscnt <= 6 instead of 0).
__device__ __forceinline__ void mm_step(const unsigned short* __restrict__ bbase,
                                        int lane, const Frag32B& a, v8f* acc) {
    Frag32B b[4];
#define LOADB(dst_, nt_)                                                     \
    do {                                                                     \
        const v4u* q_ = (const v4u*)(bbase + ((nt_)*32 + lane) * 16);        \
        (dst_).u[0]   = q_[0];                                               \
        (dst_).u[1]   = q_[1];                                               \
    } while (0)
    LOADB(b[0], 0);
    LOADB(b[1], 1);
    LOADB(b[2], 2);
#pragma unroll
    for (int nt = 0; nt < 16; ++nt) {
        if (nt + 3 < 16) LOADB(b[(nt + 3) & 3], nt + 3);
        acc[nt] = __builtin_amdgcn_wmma_f32_16x16x32_bf16(
            false, a.bf, false, b[nt & 3].bf, (short)0, acc[nt], false, false);
    }
#undef LOADB
}

// ------------------------------------------------------- WMMA GEMM + epilogue
// Block (8 waves) computes 128 rows x 256 cols. One wave = 16 rows x 256 cols
// (16 f32 accumulator tiles, 128 VGPRs). B tile double-buffered through LDS.
__global__ void __launch_bounds__(256)
k_gemm(const unsigned short* __restrict__ U,
       const unsigned short* __restrict__ WF,
       const float* __restrict__ bias,
       float* __restrict__ out, int n_nodes, int act) {
    __shared__ unsigned short Bs[2][16 * 32 * 16];   // 2 x 16 KB

    const int tid  = threadIdx.x;
    const int wid  = tid >> 5;
    const int lane = tid & 31;
    const int half = lane >> 4;
    const int m    = lane & 15;

    const long row0 = (long)blockIdx.x * 128 + (long)wid * 16;
    long rowA = row0 + m;
    if (rowA >= n_nodes) rowA = n_nodes - 1;          // clamp; stores are guarded
    const unsigned short* Arow = U + rowA * KTOT;

    v8f acc[16];
    {
        v8f z;
#pragma unroll
        for (int q = 0; q < 8; ++q) z[q] = 0.f;
#pragma unroll
        for (int t = 0; t < 16; ++t) acc[t] = z;
    }

    // ---- B-tile fill: per k-step the slice WF[kt*8192 .. +8192) ushorts
    // (= 16KB) is contiguous; 256 threads x 4 x 16B, lane-coalesced.
#if HAVE_ASYNC_LDS
#define FILL_B(kt_, buf_)                                                        \
    do {                                                                         \
        const v4u* gsrc_ = (const v4u*)(WF + (size_t)(kt_)*8192);                \
        _Pragma("unroll") for (int c_ = 0; c_ < 4; ++c_) {                       \
            int idx_ = c_ * 256 + tid;                                           \
            __builtin_amdgcn_global_load_async_to_lds_b128(                      \
                (__attribute__((address_space(1))) v4i_vs*)(gsrc_ + idx_),       \
                (__attribute__((address_space(3))) v4i_vs*)((v4u*)&Bs[buf_][0] + \
                                                            idx_),               \
                0, 0);                                                           \
        }                                                                        \
    } while (0)
#define DRAIN_B() wait_asynccnt0()
#else
    v4u pre[4];
#define FILL_B(kt_, buf_)                                                        \
    do {                                                                         \
        const v4u* gsrc_ = (const v4u*)(WF + (size_t)(kt_)*8192);                \
        _Pragma("unroll") for (int c_ = 0; c_ < 4; ++c_) pre[c_] =               \
            gsrc_[c_ * 256 + tid];                                               \
        v4u* ldst_ = (v4u*)&Bs[buf_][0];                                         \
        _Pragma("unroll") for (int c_ = 0; c_ < 4; ++c_) ldst_[c_ * 256 + tid] = \
            pre[c_];                                                             \
    } while (0)
#define DRAIN_B() ((void)0)
#endif

#define LOAD_A(kt_, frag_)                                                       \
    do {                                                                         \
        const v4u* ap_ = (const v4u*)(Arow + (kt_)*32 + half * 16);              \
        (frag_).u[0]   = ap_[0];                                                 \
        (frag_).u[1]   = ap_[1];                                                 \
    } while (0)

    // Prologue: stage kt=0, prefetch A frag for kt=0.
    FILL_B(0, 0);
    DRAIN_B();
    __syncthreads();

    Frag32B aA, aB;
    LOAD_A(0, aA);

    // k-loop manually unrolled by 2 so the in-flight A prefetch always targets
    // the register set NOT being consumed (keeps loadcnt waits non-blocking).
    for (int kt = 0; kt < KSTEPS; kt += 2) {
        // even step: compute with aA from Bs[kt&1]; prefetch kt+1 (B tile + aB)
        FILL_B(kt + 1, (kt + 1) & 1);
        LOAD_A(kt + 1, aB);
        mm_step(&Bs[kt & 1][0], lane, aA, acc);
        DRAIN_B();
        __syncthreads();

        // odd step: compute with aB; prefetch kt+2 (B tile + aA)
        if (kt + 2 < KSTEPS) {
            FILL_B(kt + 2, (kt + 2) & 1);
            LOAD_A(kt + 2, aA);
        }
        mm_step(&Bs[(kt + 1) & 1][0], lane, aB, acc);
        DRAIN_B();
        __syncthreads();
    }

    // Epilogue: C layout -> lane l, VGPR i holds C[row0 + i + 8*half][nt*16 + (l&15)]
    if (row0 < n_nodes) {
        int n = lane & 15;
#pragma unroll
        for (int nt = 0; nt < 16; ++nt) {
            int   col  = nt * 16 + n;
            float bval = bias[col];
#pragma unroll
            for (int i = 0; i < 8; ++i) {
                long row = row0 + half * 8 + i;
                if (row < n_nodes) {
                    float x = acc[nt][i] + bval;
                    x = act ? (1.0f / (1.0f + __expf(-x)))   // sigmoid (layer 2)
                            : fmaxf(x, 0.0f);                // relu    (layer 1)
                    out[row * HD + col] = x;
                }
            }
        }
    }
}

// ---------------------------------------------------------------------------
extern "C" void kernel_launch(void* const* d_in, const int* in_sizes, int n_in,
                              void* d_out, int out_size, void* d_ws, size_t ws_size,
                              hipStream_t stream) {
    const int*   node_id = (const int*)d_in[0];
    const int*   rel     = (const int*)d_in[1];
    const int*   src     = (const int*)d_in[2];
    const int*   dst     = (const int*)d_in[3];
    const float* norm    = (const float*)d_in[4];
    const float* emb     = (const float*)d_in[5];
    const float* bases1  = (const float*)d_in[6];
    const float* comp1   = (const float*)d_in[7];
    const float* loopw1  = (const float*)d_in[8];
    const float* bias1   = (const float*)d_in[9];
    const float* bases2  = (const float*)d_in[10];
    const float* comp2   = (const float*)d_in[11];
    const float* loopw2  = (const float*)d_in[12];
    const float* bias2   = (const float*)d_in[13];

    int n_nodes = in_sizes[0];
    int n_edges = in_sizes[1];

    // workspace carve-out (256B aligned): h0, h1, T(f32), U(bf16), WF(bf16)
    char*  ws  = (char*)d_ws;
    size_t off = 0;
    auto carve = [&](size_t bytes) -> void* {
        void* p = ws + off;
        off += (bytes + 255) & ~(size_t)255;
        return p;
    };
    float*          h0 = (float*)carve((size_t)n_nodes * HD * sizeof(float));
    float*          h1 = (float*)carve((size_t)n_nodes * HD * sizeof(float));
    float*          T  = (float*)carve((size_t)n_nodes * NB * HD * sizeof(float));
    unsigned short* U  = (unsigned short*)carve((size_t)n_nodes * KTOT * 2);
    unsigned short* WF = (unsigned short*)carve((size_t)KTOT * HD * 2);

    {
        long total = (long)n_nodes * HD;
        k_gather<<<(int)((total + 255) / 256), 256, 0, stream>>>(node_id, emb, h0,
                                                                 n_nodes);
    }

    const float* hcur = h0;
    for (int L = 0; L < 2; ++L) {
        const float* bases = L ? bases2 : bases1;
        const float* comp  = L ? comp2  : comp1;
        const float* loopw = L ? loopw2 : loopw1;
        const float* bias  = L ? bias2  : bias1;
        float*       hout  = L ? (float*)d_out : h1;

        long t4 = (long)n_nodes * (NB * HD) / 4;
        k_zero<<<2048, 256, 0, stream>>>((float4*)T, t4);

        k_scatter<<<(n_edges + 7) / 8, 256, 0, stream>>>(rel, src, dst, norm, hcur,
                                                         comp, T, n_edges);

        long up = (long)n_nodes * KTOT;
        k_packU<<<(int)((up + 255) / 256), 256, 0, stream>>>(T, hcur, U, n_nodes);

        k_packW<<<(KTOT * HD) / 256, 256, 0, stream>>>(bases, loopw, WF);

        int nblk = (n_nodes + 127) / 128;
        k_gemm<<<nblk, 256, 0, stream>>>(U, WF, bias, hout, n_nodes, L);

        hcur = h1;
    }
}
